// NdMamba2_2d_37374805410365
// MI455X (gfx1250) — matrix operand order repaired and checked by reading, not compiled, not run on hardware
//
#include <hip/hip_runtime.h>
#include <hip/hip_bf16.h>
#include <math.h>

// ---------------------------------------------------------------------------
// Problem constants (fixed by the reference)
// ---------------------------------------------------------------------------
constexpr int BATCH = 2;
constexpr int L     = 4096;          // H*W = 64*64
constexpr int BL    = BATCH * L;     // 8192 rows in every GEMM
constexpr int CIN   = 64;
constexpr int DM    = 256;
constexpr int COUT  = 64;
constexpr int DI    = 512;           // d_inner
constexpr int NSTATE= 16;            // d_state
constexpr int MT    = BL / 16;       // 512 M-tiles
constexpr int WPB   = 4;             // waves per GEMM block

typedef __attribute__((ext_vector_type(16))) __bf16 bf16x16;
typedef __attribute__((ext_vector_type(8)))  float  v8f;
typedef __attribute__((ext_vector_type(4)))  float  f32x4;

// ---------------------------------------------------------------------------
// Weight pack: fp32 W[N,K] row-major  ->  bf16 WMMA B-fragment layout.
// Fragment tile (nt, kt): 32 lanes x 16 bf16; lane holds N = nt*16+(lane&15),
// element j holds K = kt*32+8*(lane>>4)+(j&7)+16*(j>>3).  K zero-padded past actual K (dt_proj 16->32).
// ---------------------------------------------------------------------------
__global__ void pack_weight_bf16(const float* __restrict__ W, int N, int K,
                                 int Ktiles, __bf16* __restrict__ P, int total) {
  int tid = blockIdx.x * blockDim.x + threadIdx.x;
  if (tid >= total) return;
  int j    = tid & 15;
  int lane = (tid >> 4) & 31;
  int rem  = tid >> 9;
  int kt   = rem % Ktiles;
  int nt   = rem / Ktiles;
  int k    = kt * 32 + 8 * (lane >> 4) + (j & 7) + 16 * (j >> 3);
  int n    = nt * 16 + (lane & 15);
  float v  = (k < K) ? W[(size_t)n * K + k] : 0.0f;
  P[tid]   = (__bf16)v;
}

// ---------------------------------------------------------------------------
// Tile epilogue: C/D VGPR layout — lanes 0-15: M=v, lanes 16-31: M=v+8.
//   mode 0: plain store   mode 1: +bias, softplus (delta path)
//   mode 2: store transposed into (B, COUT, L) final output
// ---------------------------------------------------------------------------
__device__ __forceinline__ void store_tile(float* __restrict__ C, int ldc,
                                           const float* __restrict__ bias,
                                           int mode, int m0, int n0, int lane,
                                           v8f acc) {
  const int n     = n0 + (lane & 15);
  const int rbase = (lane < 16) ? 0 : 8;
#pragma unroll
  for (int v = 0; v < 8; ++v) {
    int   m   = m0 + rbase + v;
    float val = acc[v];
    if (mode == 1) {
      val += bias[n];
      val = (val > 20.0f) ? val : log1pf(__expf(val));
    }
    if (mode == 2) {
      int b = m >> 12, l = m & (L - 1);
      C[(((size_t)b * COUT + n) << 12) + l] = val;
    } else {
      C[(size_t)m * ldc + n] = val;
    }
  }
}

// ---------------------------------------------------------------------------
// WMMA bf16 GEMM:  C[M,N] = act(A[M,K] @ Wpack + bias)
// A fp32 row-major; lanes fetch their A fragment *directly* from global:
// lane holds row m=lane&15, K runs [base0, base0+8) and [base0+16, base0+24)
// — consecutive fp32, so 4 x b128 loads + packed cvt, no LDS, no masking.
// Each wave computes a 16x32 output strip (two N-tiles, 2 WMMAs / K-step).
// ---------------------------------------------------------------------------
__global__ __launch_bounds__(WPB * 32) void gemm_wmma_bf16(
    const float* __restrict__ A, int lda,
    const __bf16* __restrict__ Bp, int Ktiles, int Ntiles,
    float* __restrict__ C, int ldc,
    const float* __restrict__ bias, int mode) {
  const int lane = threadIdx.x & 31;
  const int wv   = threadIdx.x >> 5;
  const int m0   = (blockIdx.x * WPB + wv) * 16;
  const int nt0  = blockIdx.y * 2;
  const int nt1  = (nt0 + 1 < Ntiles) ? nt0 + 1 : nt0;   // clamp for odd Ntiles

  const int am    = lane & 15;
  const int base0 = (lane < 16) ? 0 : 8;

  const float*  ap  = A + (size_t)(m0 + am) * lda + base0;
  const __bf16* bp0 = Bp + ((size_t)nt0 * Ktiles * 32 + lane) * 16;
  const __bf16* bp1 = Bp + ((size_t)nt1 * Ktiles * 32 + lane) * 16;

  v8f acc0 = {}, acc1 = {};
  for (int kt = 0; kt < Ktiles; ++kt) {
    f32x4 a0 = *(const f32x4*)(ap + 0);
    f32x4 a1 = *(const f32x4*)(ap + 4);
    f32x4 a2 = *(const f32x4*)(ap + 16);
    f32x4 a3 = *(const f32x4*)(ap + 20);
    ap += 32;

    bf16x16 afrag;
#pragma unroll
    for (int c = 0; c < 4; ++c) {
      afrag[c]      = (__bf16)a0[c];   // K = base0 + 0..3
      afrag[4 + c]  = (__bf16)a1[c];   // K = base0 + 4..7
      afrag[8 + c]  = (__bf16)a2[c];   // K = base0 + 16..19
      afrag[12 + c] = (__bf16)a3[c];   // K = base0 + 20..23
    }

    bf16x16 bf0, bf1;
    ((f32x4*)&bf0)[0] = ((const f32x4*)bp0)[0];
    ((f32x4*)&bf0)[1] = ((const f32x4*)bp0)[1];
    ((f32x4*)&bf1)[0] = ((const f32x4*)bp1)[0];
    ((f32x4*)&bf1)[1] = ((const f32x4*)bp1)[1];
    if (kt + 1 < Ktiles) {
      __builtin_prefetch(bp0 + 512, 0, 0);   // global_prefetch_b8
      __builtin_prefetch(bp1 + 512, 0, 0);
    }
    bp0 += 512;
    bp1 += 512;

    acc0 = __builtin_amdgcn_wmma_f32_16x16x32_bf16(
        false, afrag, false, bf0, (short)0, acc0, false, false);
    acc1 = __builtin_amdgcn_wmma_f32_16x16x32_bf16(
        false, afrag, false, bf1, (short)0, acc1, false, false);
  }

  store_tile(C, ldc, bias, mode, m0, nt0 * 16, lane, acc0);
  if (nt1 != nt0)
    store_tile(C, ldc, bias, mode, m0, nt1 * 16, lane, acc1);
}

// ---------------------------------------------------------------------------
// (B, CIN, L) -> (B*L, CIN) row-major
// ---------------------------------------------------------------------------
__global__ void nchw_to_nlc(const float* __restrict__ x, float* __restrict__ o) {
  int tid = blockIdx.x * blockDim.x + threadIdx.x;
  if (tid >= BL * CIN) return;
  int c = tid % CIN;
  int l = (tid / CIN) % L;
  int b = tid / (CIN * L);
  o[tid] = x[((size_t)b * CIN + c) * L + l];
}

// dst[b,l,:] = src[b,L-1-l,:]
__global__ void flip_copy(const float* __restrict__ s, float* __restrict__ d, int C) {
  int tid = blockIdx.x * blockDim.x + threadIdx.x;
  if (tid >= BL * C) return;
  int c = tid % C;
  int l = (tid / C) % L;
  int b = tid / (C * L);
  d[tid] = s[((size_t)b * L + (L - 1 - l)) * C + c];
}

// o1[b,l,:] += o2[b,L-1-l,:]
__global__ void add_flip(float* __restrict__ o1, const float* __restrict__ o2, int C) {
  int tid = blockIdx.x * blockDim.x + threadIdx.x;
  if (tid >= BL * C) return;
  int c = tid % C;
  int l = (tid / C) % L;
  int b = tid / (C * L);
  o1[tid] += o2[((size_t)b * L + (L - 1 - l)) * C + c];
}

// dt columns of x_dbl -> zero-padded (BL, 32) operand for the dt_proj GEMM
__global__ void pad_dt(const float* __restrict__ xdbl, float* __restrict__ dtp) {
  int tid = blockIdx.x * blockDim.x + threadIdx.x;
  if (tid >= BL * 32) return;
  int c = tid & 31;
  int m = tid >> 5;
  dtp[tid] = (c < 16) ? xdbl[(size_t)m * 48 + c] : 0.0f;
}

// causal depthwise conv (k=4) + bias + silu; x part = xz[:, :DI]
__global__ void conv_silu(const float* __restrict__ xz, const float* __restrict__ w,
                          const float* __restrict__ cb, float* __restrict__ xc) {
  int tid = blockIdx.x * blockDim.x + threadIdx.x;
  if (tid >= BL * DI) return;
  int d = tid % DI;
  int l = (tid / DI) % L;
  int b = tid / (DI * L);
  float acc = cb[d];
#pragma unroll
  for (int j = 0; j < 4; ++j) {
    int ls = l - 3 + j;
    if (ls >= 0) acc += xz[((size_t)b * L + ls) * (2 * DI) + d] * w[d * 4 + j];
  }
  xc[tid] = acc / (1.0f + __expf(-acc));   // silu
}

// ---------------------------------------------------------------------------
// Selective scan: one block per batch, one channel per thread (DI=512).
// h[16] lives in VGPRs; B_t/C_t broadcast through LDS; gate fused.
// ---------------------------------------------------------------------------
__global__ __launch_bounds__(DI) void scan_kernel(
    const float* __restrict__ delta, const float* __restrict__ xconv,
    const float* __restrict__ xdbl,  const float* __restrict__ xz,
    const float* __restrict__ A_log, const float* __restrict__ Dp,
    float* __restrict__ ym) {
  const int d = threadIdx.x;
  const int b = blockIdx.x;
  float Arow[NSTATE];
#pragma unroll
  for (int n = 0; n < NSTATE; ++n) Arow[n] = -__expf(A_log[d * NSTATE + n]);
  float h[NSTATE];
#pragma unroll
  for (int n = 0; n < NSTATE; ++n) h[n] = 0.0f;
  const float Dd = Dp[d];

  __shared__ float sBC[2 * NSTATE];
  for (int l = 0; l < L; ++l) {
    size_t row = (size_t)b * L + l;
    if (threadIdx.x < 2 * NSTATE)
      sBC[threadIdx.x] = xdbl[row * 48 + 16 + threadIdx.x];  // B then C
    __syncthreads();
    float dlt = delta[row * DI + d];
    float u   = xconv[row * DI + d];
    float du  = dlt * u;
    float y   = 0.0f;
#pragma unroll
    for (int n = 0; n < NSTATE; ++n) {
      float dA = __expf(dlt * Arow[n]);
      h[n] = h[n] * dA + du * sBC[n];
      y += h[n] * sBC[NSTATE + n];
    }
    float z  = xz[row * (2 * DI) + DI + d];
    float sz = z / (1.0f + __expf(-z));
    ym[row * DI + d] = (y + u * Dd) * sz;
    __syncthreads();
  }
}

// ---------------------------------------------------------------------------
// Host-side orchestration
// ---------------------------------------------------------------------------
extern "C" void kernel_launch(void* const* d_in, const int* in_sizes, int n_in,
                              void* d_out, int out_size, void* d_ws, size_t ws_size,
                              hipStream_t stream) {
  (void)in_sizes; (void)n_in; (void)out_size; (void)ws_size;
  const float* x        = (const float*)d_in[0];
  const float* fc_in_w  = (const float*)d_in[1];
  const float* fc_out_w = (const float*)d_in[2];
  struct MP { const float *ipw, *cw, *cbp, *xpw, *dtw, *dtb, *alog, *Dp, *opw; };
  MP mp[2];
  for (int i = 0; i < 2; ++i) {
    int o = 3 + i * 9;
    mp[i] = { (const float*)d_in[o + 0], (const float*)d_in[o + 1],
              (const float*)d_in[o + 2], (const float*)d_in[o + 3],
              (const float*)d_in[o + 4], (const float*)d_in[o + 5],
              (const float*)d_in[o + 6], (const float*)d_in[o + 7],
              (const float*)d_in[o + 8] };
  }

  char* ws = (char*)d_ws;
  size_t off = 0;
  auto alloc = [&](size_t bytes) {
    void* p = ws + off;
    off += (bytes + 255) & ~(size_t)255;
    return p;
  };
  float* X2    = (float*)alloc((size_t)BL * CIN * 4);
  float* t     = (float*)alloc((size_t)BL * DM * 4);
  float* tf    = (float*)alloc((size_t)BL * DM * 4);
  float* xzb   = (float*)alloc((size_t)BL * 2 * DI * 4);
  float* xconv = (float*)alloc((size_t)BL * DI * 4);
  float* xdbl  = (float*)alloc((size_t)BL * 48 * 4);
  float* dtpad = (float*)alloc((size_t)BL * 32 * 4);
  float* delta = (float*)alloc((size_t)BL * DI * 4);
  float* ymb   = (float*)alloc((size_t)BL * DI * 4);
  float* o1    = (float*)alloc((size_t)BL * DM * 4);
  float* o2    = (float*)alloc((size_t)BL * DM * 4);
  // bf16 weight packs
  __bf16* p_fcin  = (__bf16*)alloc(16 * 2 * 512 * 2);
  __bf16* p_fcout = (__bf16*)alloc(4 * 8 * 512 * 2);
  __bf16* p_ip[2], *p_xp[2], *p_dt[2], *p_op[2];
  for (int i = 0; i < 2; ++i) {
    p_ip[i] = (__bf16*)alloc((size_t)64 * 8 * 512 * 2);
    p_xp[i] = (__bf16*)alloc((size_t)3 * 16 * 512 * 2);
    p_dt[i] = (__bf16*)alloc((size_t)32 * 1 * 512 * 2);
    p_op[i] = (__bf16*)alloc((size_t)16 * 16 * 512 * 2);
  }

  auto pack = [&](const float* W, int N, int K, int Kt, __bf16* P) {
    int total = (N / 16) * Kt * 512;
    pack_weight_bf16<<<(total + 255) / 256, 256, 0, stream>>>(W, N, K, Kt, P, total);
  };
  auto gemm = [&](const float* A, int lda, const __bf16* P, int Kt,
                  int Ntiles, float* C, int ldc, const float* bias, int mode) {
    gemm_wmma_bf16<<<dim3(MT / WPB, (Ntiles + 1) / 2), WPB * 32, 0, stream>>>(
        A, lda, P, Kt, Ntiles, C, ldc, bias, mode);
  };
  auto ew = [&](int total) { return dim3((total + 255) / 256); };

  // ---- weight packing (cheap, every launch: deterministic, no caching) ----
  pack(fc_in_w,  DM,   CIN, 2,  p_fcin);
  pack(fc_out_w, COUT, DM,  8,  p_fcout);
  for (int i = 0; i < 2; ++i) {
    pack(mp[i].ipw, 2 * DI, DM, 8,  p_ip[i]);   // (1024,256)
    pack(mp[i].xpw, 48,     DI, 16, p_xp[i]);   // (48,512)
    pack(mp[i].dtw, DI,     16, 1,  p_dt[i]);   // (512,16) K padded to 32
    pack(mp[i].opw, DM,     DI, 16, p_op[i]);   // (256,512)
  }

  // ---- fc_in: (B,CIN,L) -> (BL,CIN) -> t (BL,DM) ----
  nchw_to_nlc<<<ew(BL * CIN), 256, 0, stream>>>(x, X2);
  gemm(X2, CIN, p_fcin, 2, DM / 16, t, DM, nullptr, 0);
  flip_copy<<<ew(BL * DM), 256, 0, stream>>>(t, tf, DM);

  // ---- two mamba passes (forward on t, backward on tf) ----
  for (int i = 0; i < 2; ++i) {
    const float* u = (i == 0) ? t : tf;
    // in_proj: xz = u @ W^T  (BL,1024)
    gemm(u, DM, p_ip[i], 8, (2 * DI) / 16, xzb, 2 * DI, nullptr, 0);
    // causal depthwise conv + silu -> xconv (BL,512)
    conv_silu<<<ew(BL * DI), 256, 0, stream>>>(xzb, mp[i].cw, mp[i].cbp, xconv);
    // x_proj: x_dbl = xconv @ W^T  (BL,48)
    gemm(xconv, DI, p_xp[i], 16, 48 / 16, xdbl, 48, nullptr, 0);
    // dt_proj + bias + softplus: delta = softplus(dt @ W^T + b)  (BL,512)
    pad_dt<<<ew(BL * 32), 256, 0, stream>>>(xdbl, dtpad);
    gemm(dtpad, 32, p_dt[i], 1, DI / 16, delta, DI, mp[i].dtb, 1);
    // selective scan + D skip + silu(z) gate -> ym (BL,512)
    scan_kernel<<<BATCH, DI, 0, stream>>>(delta, xconv, xdbl, xzb,
                                          mp[i].alog, mp[i].Dp, ymb);
    // out_proj -> o (BL,256)
    gemm(ymb, DI, p_op[i], 16, DM / 16, (i == 0) ? o1 : o2, DM, nullptr, 0);
  }

  // ---- combine: o1 += flip(o2); final fc_out with fused NCHW transpose ----
  add_flip<<<ew(BL * DM), 256, 0, stream>>>(o1, o2, DM);
  gemm(o1, DM, p_fcout, 8, COUT / 16, (float*)d_out, COUT, nullptr, 2);
}
